// Decoder_82952998355416
// MI455X (gfx1250) — compile-verified, matrix operand order checked by reading, not run on hardware
//
#include <hip/hip_runtime.h>

// ---------------------------------------------------------------------------
// Fused graph-LSTM decoder for MI455X (gfx1250), wave32 + WMMA bf16.
// One persistent kernel: 64 blocks x 256 threads; block = 8 batch rows,
// all 24 nodes, all 25 timesteps. h-state in LDS (bf16), c-state in L2
// scratch, weights pre-packed to WMMA B-operand layout (bf16).
// ---------------------------------------------------------------------------

typedef __attribute__((ext_vector_type(16))) __bf16 v16bf;
typedef __attribute__((ext_vector_type(8)))  float  v8f;

union BF16x16 { v16bf v; uint4 u[2]; };
union F32x8   { v8f  v; uint4 u[2]; float f[8]; };

#define DEVINL __device__ __forceinline__

constexpr int Bb   = 512;   // batch
constexpr int Nn   = 24;    // nodes
constexpr int Tt   = 25;    // timesteps
constexpr int BT   = 8;     // batch rows per block (rows 8..15 of WMMA M duplicated)
constexpr int NBLK = Bb / BT;  // 64 blocks
constexpr int NPW  = 3;     // nodes per wave (8 waves * 3 = 24)
constexpr int KX   = 96;    // xi K padded (64 z + 8 dyn + 24 zero)

// ---- workspace layout (bytes; all sizes multiples of 1KB -> 16B aligned) ----
constexpr size_t SZ_WX0  = (size_t)Nn*32*3*512*2;   // packed Wx0 (K=96)
constexpr size_t SZ_WH   = (size_t)Nn*32*4*512*2;   // packed Wh0/Wx1/Wh1 (K=128)
constexpr size_t SZ_FCW  = (size_t)Nn*8*4*512*2;    // packed fc/fc2 (128x128)
constexpr size_t SZ_IHW  = (size_t)8*8*512*2;       // packed ih (256x128), n-shared
constexpr size_t SZ_HDW  = (size_t)4*512*2;         // packed head (128x16pad)
constexpr size_t SZ_B512 = (size_t)Nn*32*256*4;     // C-layout bias, 512 cols
constexpr size_t SZ_B128 = (size_t)Nn*8*256*4;      // C-layout bias, 128 cols
constexpr size_t SZ_IHB  = (size_t)8*256*4;
constexpr size_t SZ_HDB  = (size_t)256*4;
constexpr size_t SZ_C    = (size_t)NBLK*Nn*8*256*4; // c state, tile-packed

constexpr size_t OFF_WX0   = 0;
constexpr size_t OFF_WH0   = OFF_WX0 + SZ_WX0;
constexpr size_t OFF_WX1   = OFF_WH0 + SZ_WH;
constexpr size_t OFF_WH1   = OFF_WX1 + SZ_WH;
constexpr size_t OFF_FCW   = OFF_WH1 + SZ_WH;
constexpr size_t OFF_FC2W  = OFF_FCW + SZ_FCW;
constexpr size_t OFF_IH1W  = OFF_FC2W + SZ_FCW;
constexpr size_t OFF_IH2W  = OFF_IH1W + SZ_IHW;
constexpr size_t OFF_LOCW  = OFF_IH2W + SZ_IHW;
constexpr size_t OFF_LOGZW = OFF_LOCW + SZ_HDW;
constexpr size_t OFF_B0    = OFF_LOGZW + SZ_HDW;
constexpr size_t OFF_B1    = OFF_B0 + SZ_B512;
constexpr size_t OFF_FCB   = OFF_B1 + SZ_B512;
constexpr size_t OFF_FC2B  = OFF_FCB + SZ_B128;
constexpr size_t OFF_IH1B  = OFF_FC2B + SZ_B128;
constexpr size_t OFF_IH2B  = OFF_IH1B + SZ_IHB;
constexpr size_t OFF_LOCB  = OFF_IH2B + SZ_IHB;
constexpr size_t OFF_LOGZB = OFF_LOCB + SZ_HDB;
constexpr size_t OFF_C1    = OFF_LOGZB + SZ_HDB;
constexpr size_t OFF_C2    = OFF_C1 + SZ_C;

// ---- LDS layout (bytes) -----------------------------------------------------
constexpr int LDS_X     = 0;                       // [24][8][96] bf16   36,864
constexpr int LDS_H1    = 36864;                   // 2 x [24][8][128]   98,304
constexpr int LDS_H2    = 135168;                  // 2 x [24][8][128]   98,304
constexpr int LDS_Y2    = 233472;                  // [24][8][128] bf16  49,152
constexpr int LDS_S     = 282624;                  // [24][32][8] f32    24,576
constexpr int LDS_G     = 307200;                  // [24][24] f32        2,304
constexpr int LDS_LOCST = 309504;                  // [24][8][4] f32      3,072
constexpr int LDS_HEADA = 312576;                  // [24][8][4] f32      3,072
constexpr int LDS_HEADB = 315648;                  // [24][8][4] f32      3,072
constexpr int LDS_TOTAL = 318720;                  // <= 320KB WGP LDS

DEVINL unsigned short f2bf(float x) {
  union { float f; unsigned u; } v; v.f = x;
  unsigned r = v.u + 0x7FFFu + ((v.u >> 16) & 1u);   // RNE
  return (unsigned short)(r >> 16);
}
DEVINL float sigm(float x) { return 1.0f / (1.0f + __expf(-x)); }

// A operand: node base = [8 rows][Kpad] bf16; lane l covers row m=l%16 (rows
// 8..15 duplicate 0..7 since only 8 batch rows are real), halves follow the
// ISA 16-bit A layout: k = kc*32 + {g*8..g*8+7, 16+g*8..16+g*8+7}.
DEVINL v16bf load_a_tile(const unsigned short* nodeBase, int Kpad, int m, int g, int kc) {
  int row = m & 7;
  const unsigned short* p = nodeBase + row * Kpad + kc * 32 + g * 8;
  BF16x16 r;
  r.u[0] = *(const uint4*)(p);
  r.u[1] = *(const uint4*)(p + 16);
  return r.v;
}
// B operand: pre-packed so each lane reads 16 contiguous halves (32B).
DEVINL v16bf load_b_tile(const unsigned short* packed, int tile, int lane) {
  const uint4* p = (const uint4*)(packed + (size_t)tile * 512 + lane * 16);
  BF16x16 r; r.u[0] = p[0]; r.u[1] = p[1];
  return r.v;
}
DEVINL v8f load_c8(const float* p) {
  const uint4* q = (const uint4*)p;
  F32x8 r; r.u[0] = q[0]; r.u[1] = q[1];
  return r.v;
}
DEVINL void store_c8(float* p, v8f v) {
  F32x8 r; r.v = v;
  uint4* q = (uint4*)p; q[0] = r.u[0]; q[1] = r.u[1];
}

DEVINL v8f gemm_tiles(v8f acc, const unsigned short* shA, int Kpad,
                      const unsigned short* Bp, int tb, int KC, int m, int g, int lane) {
#pragma unroll
  for (int kc = 0; kc < KC; ++kc) {
    v16bf a = load_a_tile(shA, Kpad, m, g, kc);
    v16bf b = load_b_tile(Bp, tb + kc, lane);
    acc = __builtin_amdgcn_wmma_f32_16x16x32_bf16(false, a, false, b, (short)0, acc,
                                                  false, false);
  }
  return acc;
}

// One mix unit: per wave, 3 nodes: acc = bias + A1*B1 (+ A2*B2) -> S (LDS),
// barrier, then mixed[m] = sum_n G[m,n] * S[n] for the wave's 3 output nodes.
DEVINL void mix_unit(v8f out[NPW], float* S, const float* shG, int wv, int lane, int ot,
                     const float* Cp, int cOT, bool cPerN,
                     const unsigned short* A1, int Kpad1, const unsigned short* B1,
                     int KC1, int bOT1, bool b1PerN,
                     const unsigned short* A2, int Kpad2, const unsigned short* B2,
                     int KC2, int bOT2, bool b2PerN) {
  const int m = lane & 15, g = lane >> 4;
#pragma unroll
  for (int j = 0; j < NPW; ++j) {
    int n = wv * NPW + j;
    v8f acc = load_c8(Cp + ((size_t)(cPerN ? n : 0) * cOT + ot) * 256 + lane * 8);
    acc = gemm_tiles(acc, A1 + (size_t)n * 8 * Kpad1, Kpad1, B1,
                     ((b1PerN ? n : 0) * bOT1 + ot) * KC1, KC1, m, g, lane);
    if (A2)
      acc = gemm_tiles(acc, A2 + (size_t)n * 8 * Kpad2, Kpad2, B2,
                       ((b2PerN ? n : 0) * bOT2 + ot) * KC2, KC2, m, g, lane);
    store_c8(S + n * 256 + lane * 8, acc);
  }
  __syncthreads();
#pragma unroll
  for (int j = 0; j < NPW; ++j) {
    int mm = wv * NPW + j;
    F32x8 acc;
#pragma unroll
    for (int e = 0; e < 8; ++e) acc.f[e] = 0.f;
    for (int n = 0; n < Nn; ++n) {
      float gg = shG[mm * Nn + n];
      F32x8 s;
      const uint4* p = (const uint4*)(S + n * 256 + lane * 8);
      s.u[0] = p[0]; s.u[1] = p[1];
      acc.v += s.v * gg;   // VALU fma co-executes with WMMA (TRANS)
    }
    out[j] = acc.v;
  }
  __syncthreads();
}

DEVINL void lstm_layer(int bt, float* cState, unsigned short* shHnew, unsigned short* shYi,
                       float* S, const float* shG, int wv, int lane, const float* biasP,
                       const unsigned short* A1, int Kpad1, const unsigned short* B1, int KC1,
                       const unsigned short* A2, int Kpad2, const unsigned short* B2, int KC2) {
#pragma unroll 1
  for (int dt = 0; dt < 8; ++dt) {
    v8f gate[4][NPW];
#pragma unroll
    for (int gi = 0; gi < 4; ++gi) {
      v8f o[NPW];
      mix_unit(o, S, shG, wv, lane, gi * 8 + dt, biasP, 32, true,
               A1, Kpad1, B1, KC1, 32, true, A2, Kpad2, B2, KC2, 32, true);
#pragma unroll
      for (int j = 0; j < NPW; ++j) gate[gi][j] = o[j];
    }
#pragma unroll
    for (int j = 0; j < NPW; ++j) {
      int m = wv * NPW + j;
      float* cptr = cState + (((size_t)bt * Nn + m) * 8 + dt) * 256 + lane * 8;
      F32x8 cold;
      { const uint4* p = (const uint4*)cptr; cold.u[0] = p[0]; cold.u[1] = p[1]; }
      F32x8 gI, gF, gG, gO, cnew;
      gI.v = gate[0][j]; gF.v = gate[1][j]; gG.v = gate[2][j]; gO.v = gate[3][j];
      float hval[8];
#pragma unroll
      for (int e = 0; e < 8; ++e) {
        float cn = sigm(gF.f[e]) * cold.f[e] + sigm(gI.f[e]) * tanhf(gG.f[e]);
        cnew.f[e] = cn;
        hval[e] = sigm(gO.f[e]) * tanhf(cn);
      }
      { uint4* p = (uint4*)cptr; p[0] = cnew.u[0]; p[1] = cnew.u[1]; }
      if (lane < 16) {   // rows 0..7 live in lanes 0..15
        int ch = dt * 16 + lane;
#pragma unroll
        for (int e = 0; e < 8; ++e) {
          shHnew[(m * 8 + e) * 128 + ch] = f2bf(hval[e]);
          if (shYi) shYi[(m * 8 + e) * 128 + ch] = f2bf(tanhf(hval[e]));
        }
      }
    }
  }
}

__global__ __launch_bounds__(256)
void decoder_main(const float* __restrict__ xin, const float* __restrict__ enc,
                  const float* __restrict__ zin, const float* __restrict__ Gin,
                  unsigned char* __restrict__ ws, float* __restrict__ dout) {
  __shared__ __align__(16) unsigned char smem[LDS_TOTAL];
  const int tid = threadIdx.x, wv = tid >> 5, lane = tid & 31, bt = blockIdx.x;

  unsigned short* shX  = (unsigned short*)(smem + LDS_X);
  unsigned short* shH1 = (unsigned short*)(smem + LDS_H1);
  unsigned short* shH2 = (unsigned short*)(smem + LDS_H2);
  unsigned short* shY2 = (unsigned short*)(smem + LDS_Y2);
  float* S     = (float*)(smem + LDS_S);
  float* shG   = (float*)(smem + LDS_G);
  float* shLoc = (float*)(smem + LDS_LOCST);
  float* headA = (float*)(smem + LDS_HEADA);
  float* headB = (float*)(smem + LDS_HEADB);
  unsigned short* shEnc = shH1 + 24576;  // aliases H1[1]+H2[0] (dead until step0 writes)

  const unsigned short* Wx0p  = (const unsigned short*)(ws + OFF_WX0);
  const unsigned short* Wh0p  = (const unsigned short*)(ws + OFF_WH0);
  const unsigned short* Wx1p  = (const unsigned short*)(ws + OFF_WX1);
  const unsigned short* Wh1p  = (const unsigned short*)(ws + OFF_WH1);
  const unsigned short* fcWp  = (const unsigned short*)(ws + OFF_FCW);
  const unsigned short* fc2Wp = (const unsigned short*)(ws + OFF_FC2W);
  const unsigned short* ih1Wp = (const unsigned short*)(ws + OFF_IH1W);
  const unsigned short* ih2Wp = (const unsigned short*)(ws + OFF_IH2W);
  const unsigned short* locWp = (const unsigned short*)(ws + OFF_LOCW);
  const unsigned short* lzWp  = (const unsigned short*)(ws + OFF_LOGZW);
  const float* b0p   = (const float*)(ws + OFF_B0);
  const float* b1p   = (const float*)(ws + OFF_B1);
  const float* fcbp  = (const float*)(ws + OFF_FCB);
  const float* fc2bp = (const float*)(ws + OFF_FC2B);
  const float* ih1bp = (const float*)(ws + OFF_IH1B);
  const float* ih2bp = (const float*)(ws + OFF_IH2B);
  const float* locbp = (const float*)(ws + OFF_LOCB);
  const float* lzbp  = (const float*)(ws + OFF_LOGZB);
  float* c1p = (float*)(ws + OFF_C1);
  float* c2p = (float*)(ws + OFF_C2);

  // ---------------- init: G, xi (z + x dyn), loc_start, enc -> LDS ----------
  for (int i = tid; i < Nn * Nn; i += 256) shG[i] = Gin[i];
  for (int i = tid; i < Nn * BT; i += 256) {
    int n = i / BT, b = i % BT, bg = bt * BT + b;
    const float* zp = zin + ((size_t)bg * Nn + n) * 64;
    const float* xp = xin + ((size_t)bg * Nn + n) * 8;
    unsigned short* d = shX + (n * BT + b) * KX;
    for (int k = 0; k < 64; ++k) d[k] = f2bf(zp[k]);
    for (int k = 0; k < 8;  ++k) d[64 + k] = f2bf(xp[k]);
    for (int k = 72; k < KX; ++k) d[k] = 0;
    float* ls = shLoc + (n * BT + b) * 4;
    for (int c = 0; c < 4; ++c) ls[c] = xp[c];
  }
  for (int i = tid; i < Nn * BT * 256; i += 256) {
    int k = i & 255, nb = i >> 8;
    int n = nb / BT, b = nb % BT, bg = bt * BT + b;
    shEnc[i] = f2bf(enc[((size_t)bg * Nn + n) * 256 + k]);
  }
  __syncthreads();

  // ---------------- h0 / c0 (graph_linear on enc) ---------------------------
#pragma unroll 1
  for (int ot = 0; ot < 8; ++ot) {
    v8f h0[NPW], c0[NPW];
    mix_unit(h0, S, shG, wv, lane, ot, ih1bp, 8, false,
             shEnc, 256, ih1Wp, 8, 8, false, nullptr, 0, nullptr, 0, 0, false);
    mix_unit(c0, S, shG, wv, lane, ot, ih2bp, 8, false,
             shEnc, 256, ih2Wp, 8, 8, false, nullptr, 0, nullptr, 0, 0, false);
#pragma unroll
    for (int j = 0; j < NPW; ++j) {
      int m = wv * NPW + j;
      float* p1 = c1p + (((size_t)bt * Nn + m) * 8 + ot) * 256 + lane * 8;
      float* p2 = c2p + (((size_t)bt * Nn + m) * 8 + ot) * 256 + lane * 8;
      store_c8(p1, c0[j]);
      store_c8(p2, c0[j]);
      if (lane < 16) {
        int ch = ot * 16 + lane;
        F32x8 h; h.v = h0[j];
#pragma unroll
        for (int e = 0; e < 8; ++e) {
          unsigned short hb = f2bf(h.f[e]);
          shH1[(m * 8 + e) * 128 + ch] = hb;             // h1 buf0
          (shH2 + 24576)[(m * 8 + e) * 128 + ch] = hb;   // h2 buf1
        }
      }
    }
  }
  __syncthreads();

  // ---------------- T recurrent steps ---------------------------------------
#pragma unroll 1
  for (int t = 0; t < Tt; ++t) {
    const int p = t & 1;
    unsigned short* h1r = shH1 + p * 24576;
    unsigned short* h1w = shH1 + (1 - p) * 24576;
    unsigned short* h2r = shH2 + (1 - p) * 24576;
    unsigned short* h2w = shH2 + p * 24576;
    unsigned short* yi  = h1r;   // free after layer1 consumes it
    unsigned short* yi1 = h2r;   // free after layer2 consumes it

    lstm_layer(bt, c1p, h1w, nullptr, S, shG, wv, lane, b0p,
               shX, KX, Wx0p, 3, h1r, 128, Wh0p, 4);
    __syncthreads();
    lstm_layer(bt, c2p, h2w, yi, S, shG, wv, lane, b1p,
               h1w, 128, Wx1p, 4, h2r, 128, Wh1p, 4);
    __syncthreads();

    // fc / fc2: yi1 = tanh(mix(yi@fcW+b)), yi2 = tanh(mix(yi@fc2W+b))
#pragma unroll 1
    for (int ot = 0; ot < 8; ++ot) {
      v8f o1[NPW], o2[NPW];
      mix_unit(o1, S, shG, wv, lane, ot, fcbp, 8, true,
               yi, 128, fcWp, 4, 8, true, nullptr, 0, nullptr, 0, 0, false);
#pragma unroll
      for (int j = 0; j < NPW; ++j) {
        int m = wv * NPW + j;
        if (lane < 16) {
          int ch = ot * 16 + lane;
          F32x8 v; v.v = o1[j];
#pragma unroll
          for (int e = 0; e < 8; ++e) yi1[(m * 8 + e) * 128 + ch] = f2bf(tanhf(v.f[e]));
        }
      }
      mix_unit(o2, S, shG, wv, lane, ot, fc2bp, 8, true,
               yi, 128, fc2Wp, 4, 8, true, nullptr, 0, nullptr, 0, 0, false);
#pragma unroll
      for (int j = 0; j < NPW; ++j) {
        int m = wv * NPW + j;
        if (lane < 16) {
          int ch = ot * 16 + lane;
          F32x8 v; v.v = o2[j];
#pragma unroll
          for (int e = 0; e < 8; ++e) shY2[(m * 8 + e) * 128 + ch] = f2bf(tanhf(v.f[e]));
        }
      }
    }
    __syncthreads();

    // heads: loc3 = mix(yi1@locW+b), logz = mix(yi2@logzW+b)  (O padded to 16)
    {
      v8f l3[NPW], lz[NPW];
      mix_unit(l3, S, shG, wv, lane, 0, locbp, 1, false,
               yi1, 128, locWp, 4, 1, false, nullptr, 0, nullptr, 0, 0, false);
#pragma unroll
      for (int j = 0; j < NPW; ++j) {
        int m = wv * NPW + j;
        if (lane < 4) {
          F32x8 v; v.v = l3[j];
#pragma unroll
          for (int e = 0; e < 8; ++e) headA[(m * 8 + e) * 4 + lane] = v.f[e];
        }
      }
      mix_unit(lz, S, shG, wv, lane, 0, lzbp, 1, false,
               shY2, 128, lzWp, 4, 1, false, nullptr, 0, nullptr, 0, 0, false);
#pragma unroll
      for (int j = 0; j < NPW; ++j) {
        int m = wv * NPW + j;
        if (lane < 4) {
          F32x8 v; v.v = lz[j];
#pragma unroll
          for (int e = 0; e < 8; ++e) headB[(m * 8 + e) * 4 + lane] = v.f[e];
        }
      }
    }
    __syncthreads();

    // quaternion update + outputs + next xi dynamic channels
    if (tid < Nn * BT) {
      int m = tid / BT, b = tid % BT, bg = bt * BT + b;
      float a0 = headA[tid * 4 + 0], a1 = headA[tid * 4 + 1], a2 = headA[tid * 4 + 2];
      float rs = rsqrtf(1.f + a0 * a0 + a1 * a1 + a2 * a2);
      float dw = rs, dx = a0 * rs, dy = a1 * rs, dz = a2 * rs;
      float* ls = shLoc + tid * 4;
      float w1 = ls[0], x1 = ls[1], y1 = ls[2], z1 = ls[3];
      float ow = w1 * dw - x1 * dx - y1 * dy - z1 * dz;
      float ox = w1 * dx + x1 * dw + y1 * dz - z1 * dy;
      float oy = w1 * dy - x1 * dz + y1 * dw + z1 * dx;
      float oz = w1 * dz + x1 * dy - y1 * dx + z1 * dw;
      size_t o1 = (((size_t)bg * Tt + t) * Nn + m) * 4;
      dout[o1 + 0] = ow; dout[o1 + 1] = ox; dout[o1 + 2] = oy; dout[o1 + 3] = oz;
      size_t o2 = (size_t)Bb * Tt * Nn * 4 + o1;
      dout[o2 + 0] = headB[tid * 4 + 0]; dout[o2 + 1] = headB[tid * 4 + 1];
      dout[o2 + 2] = headB[tid * 4 + 2]; dout[o2 + 3] = headB[tid * 4 + 3];
      ls[0] = ow; ls[1] = ox; ls[2] = oy; ls[3] = oz;
      unsigned short* xp = shX + (m * BT + b) * KX + 64;
      xp[0] = f2bf(ow); xp[1] = f2bf(ox); xp[2] = f2bf(oy); xp[3] = f2bf(oz);
      xp[4] = f2bf(dw); xp[5] = f2bf(dx); xp[6] = f2bf(dy); xp[7] = f2bf(dz);
    }
    __syncthreads();
  }
}

// ---- weight pre-pack: f32 [n][K][O] -> bf16 WMMA B-operand layout -----------
__global__ void pack_b_kernel(const float* __restrict__ W, unsigned short* __restrict__ dst,
                              int nN, int Ksrc, int Osrc, int KC, int OT, int Klim,
                              int Olim, long total) {
  long idx = (long)blockIdx.x * blockDim.x + threadIdx.x;
  if (idx >= total) return;
  int j = (int)(idx & 15);
  int lane = (int)((idx >> 4) & 31);
  long r = idx >> 9;
  int kc = (int)(r % KC); r /= KC;
  int ot = (int)(r % OT);
  int n  = (int)(r / OT);
  int c = lane & 15, g = lane >> 4;
  int vg = j >> 1, pp = j & 1;
  int klocal = ((vg < 4) ? 0 : 16) + ((vg & 3) << 1) + pp + g * 8;
  int k = kc * 32 + klocal;
  int o = ot * 16 + c;
  float val = 0.f;
  if (k < Klim && o < Olim) val = W[((long)n * Ksrc + k) * Osrc + o];
  dst[idx] = f2bf(val);
}

// ---- bias pre-pack: f32 [n][O] -> f32 WMMA C-operand tile layout ------------
__global__ void pack_c_kernel(const float* __restrict__ bsrc, float* __restrict__ dst,
                              int nN, int Osrc, int OT, int Olim, long total) {
  long idx = (long)blockIdx.x * blockDim.x + threadIdx.x;
  if (idx >= total) return;
  int lane = (int)((idx >> 3) & 31);
  long r = idx >> 8;
  int ot = (int)(r % OT);
  int n  = (int)(r / OT);
  int c = lane & 15;
  int o = ot * 16 + c;
  dst[idx] = (o < Olim) ? bsrc[(long)n * Osrc + o] : 0.f;
}

extern "C" void kernel_launch(void* const* d_in, const int* in_sizes, int n_in,
                              void* d_out, int out_size, void* d_ws, size_t ws_size,
                              hipStream_t stream) {
  (void)in_sizes; (void)n_in; (void)out_size; (void)ws_size;
  const float* x    = (const float*)d_in[0];
  const float* enc  = (const float*)d_in[1];
  const float* z    = (const float*)d_in[2];
  /* d_in[3] = y (unused by reference) */
  const float* G    = (const float*)d_in[4];
  const float* Wx0  = (const float*)d_in[5];
  const float* Wh0  = (const float*)d_in[6];
  const float* b0   = (const float*)d_in[7];
  const float* Wx1  = (const float*)d_in[8];
  const float* Wh1  = (const float*)d_in[9];
  const float* b1   = (const float*)d_in[10];
  const float* fcW  = (const float*)d_in[11];
  const float* fcb  = (const float*)d_in[12];
  const float* fc2W = (const float*)d_in[13];
  const float* fc2b = (const float*)d_in[14];
  const float* ih1W = (const float*)d_in[15];
  const float* ih1b = (const float*)d_in[16];
  const float* ih2W = (const float*)d_in[17];
  const float* ih2b = (const float*)d_in[18];
  const float* locW = (const float*)d_in[19];
  const float* locb = (const float*)d_in[20];
  const float* lzW  = (const float*)d_in[21];
  const float* lzb  = (const float*)d_in[22];
  unsigned char* ws = (unsigned char*)d_ws;
  float* out = (float*)d_out;

  auto PB = [&](const float* W, size_t off, int nN, int Ksrc, int Osrc, int KC, int OT,
                int Klim, int Olim) {
    long total = (long)nN * OT * KC * 512;
    pack_b_kernel<<<dim3((unsigned)((total + 255) / 256)), dim3(256), 0, stream>>>(
        W, (unsigned short*)(ws + off), nN, Ksrc, Osrc, KC, OT, Klim, Olim, total);
  };
  auto PC = [&](const float* b, size_t off, int nN, int Osrc, int OT, int Olim) {
    long total = (long)nN * OT * 256;
    pack_c_kernel<<<dim3((unsigned)((total + 255) / 256)), dim3(256), 0, stream>>>(
        b, (float*)(ws + off), nN, Osrc, OT, Olim, total);
  };

  PB(Wx0,  OFF_WX0,  24, 72, 512, 3, 32, 72, 512);
  PB(Wh0,  OFF_WH0,  24, 128, 512, 4, 32, 128, 512);
  PB(Wx1,  OFF_WX1,  24, 128, 512, 4, 32, 128, 512);
  PB(Wh1,  OFF_WH1,  24, 128, 512, 4, 32, 128, 512);
  PB(fcW,  OFF_FCW,  24, 128, 128, 4, 8, 128, 128);
  PB(fc2W, OFF_FC2W, 24, 128, 128, 4, 8, 128, 128);
  PB(ih1W, OFF_IH1W, 1, 256, 128, 8, 8, 256, 128);
  PB(ih2W, OFF_IH2W, 1, 256, 128, 8, 8, 256, 128);
  PB(locW, OFF_LOCW, 1, 128, 3, 4, 1, 128, 3);
  PB(lzW,  OFF_LOGZW, 1, 128, 4, 4, 1, 128, 4);
  PC(b0,   OFF_B0,   24, 512, 32, 512);
  PC(b1,   OFF_B1,   24, 512, 32, 512);
  PC(fcb,  OFF_FCB,  24, 128, 8, 128);
  PC(fc2b, OFF_FC2B, 24, 128, 8, 128);
  PC(ih1b, OFF_IH1B, 1, 128, 8, 128);
  PC(ih2b, OFF_IH2B, 1, 128, 8, 128);
  PC(locb, OFF_LOCB, 1, 3, 1, 3);
  PC(lzb,  OFF_LOGZB, 1, 4, 1, 4);

  decoder_main<<<dim3(NBLK), dim3(256), 0, stream>>>(x, enc, z, G, ws, out);
}
